// MoELayerWithSkip_27608049779044
// MI455X (gfx1250) — matrix operand order repaired and checked by reading, not compiled, not run on hardware
//
#include <hip/hip_runtime.h>
#include <hip/hip_bf16.h>
#include <cstdint>

#define N_TOK 8192
#define DIM   1024
#define HID   4096
#define NE    8
#define XN_STR 1032   // 1024 + 8 pad (dword stride % 64 == 4 -> conflict-free b128 A loads)
#define HC_STR 520    // 512 + 8 pad  (dword stride % 64 == 4)
#define HCHUNK 512

typedef __attribute__((ext_vector_type(16))) __bf16 v16bf;
typedef __attribute__((ext_vector_type(8)))  float  v8f;

// ---------------------------------------------------------------- utilities
__device__ __forceinline__ float block_reduce_sum(float v, volatile float* red) {
    int tid = threadIdx.x, lane = tid & 31, w = tid >> 5;
#pragma unroll
    for (int o = 16; o; o >>= 1) v += __shfl_xor(v, o, 32);
    __syncthreads();                 // protect red reuse from a previous call
    if (lane == 0) red[w] = v;
    __syncthreads();
    float r = 0.f;
#pragma unroll
    for (int i = 0; i < 8; ++i) r += red[i];
    return r;
}

// A fragment (16x32 bf16) from an LDS row-major tile. Contiguous K (matches B):
// lane half 0: K = k0 + {0..15}; lane half 1: K = k0 + {16..31}
__device__ __forceinline__ v16bf load_a_frag(const __bf16* base, int stride, int k0) {
    int lane = threadIdx.x & 31;
    int m = lane & 15, half = lane >> 4;
    const __bf16* p = base + m * stride + k0 + half * 16;
    union { uint4 u[2]; v16bf v; } r;
    r.u[0] = *(const uint4*)(p);        // elements 0..7
    r.u[1] = *(const uint4*)(p + 8);    // elements 8..15
    return r.v;
}

// B fragment (32x16) gathered from row-major fp32 weights W[K][N] (leading dim ld),
// converted to bf16 in-register. Contiguous K (matches A): N = lane%16, K = k0 + 16*(lane/16) + i.
__device__ __forceinline__ v16bf load_b_frag_global(const float* __restrict__ w,
                                                    int ld, int k0, int ncol) {
    int lane = threadIdx.x & 31;
    int n = lane & 15, kh = (lane >> 4) << 4;
    const float* p = w + (size_t)(k0 + kh) * ld + ncol + n;
    v16bf b;
#pragma unroll
    for (int i = 0; i < 16; ++i) b[i] = (__bf16)p[(size_t)i * ld];
    return b;
}

__device__ __forceinline__ v8f wmma_bf16(v16bf a, v16bf b, v8f c) {
    return __builtin_amdgcn_wmma_f32_16x16x32_bf16(false, a, false, b,
                                                   (short)0, c, false, false);
}

// ------------------------------------------------- kernel 1: LN + router
__global__ __launch_bounds__(256) void ln_router_kernel(
    const float* __restrict__ x, const float* __restrict__ ln_g,
    const float* __restrict__ ln_b, const float* __restrict__ conf_w,
    const float* __restrict__ conf_b, const float* __restrict__ router_w,
    const float* __restrict__ router_b,
    __bf16* __restrict__ xn_ws, float* __restrict__ cw,
    int* __restrict__ lists, int* __restrict__ counts,
    float* __restrict__ out_main, float* __restrict__ out_conf,
    float* __restrict__ out_xcopy)
{
    __shared__ float red[8];
    __shared__ float redE[64];
    __shared__ float rwt[NE * DIM];   // router weights transposed [e][d]

    int tok = blockIdx.x, tid = threadIdx.x;
    int lane = tid & 31, w = tid >> 5;

    // stage router weights transposed (coalesced global reads)
    for (int idx = tid; idx < DIM * NE; idx += 256) {
        int j = idx >> 3, e = idx & 7;
        rwt[e * DIM + j] = router_w[idx];
    }

    const float4 xv = ((const float4*)(x + (size_t)tok * DIM))[tid];
    float s = xv.x + xv.y + xv.z + xv.w;
    float mean = block_reduce_sum(s, red) * (1.f / DIM);   // barrier also covers rwt staging
    float d0 = xv.x - mean, d1 = xv.y - mean, d2 = xv.z - mean, d3 = xv.w - mean;
    float var = block_reduce_sum(d0*d0 + d1*d1 + d2*d2 + d3*d3, red) * (1.f / DIM);
    float rstd = rsqrtf(var + 1e-5f);

    float4 gv = ((const float4*)ln_g)[tid];
    float4 bv = ((const float4*)ln_b)[tid];
    float xn0 = d0 * rstd * gv.x + bv.x;
    float xn1 = d1 * rstd * gv.y + bv.y;
    float xn2 = d2 * rstd * gv.z + bv.z;
    float xn3 = d3 * rstd * gv.w + bv.w;

    { // store x_norm as bf16
        union { __bf16 h[4]; uint2 u; } pk;
        pk.h[0] = (__bf16)xn0; pk.h[1] = (__bf16)xn1;
        pk.h[2] = (__bf16)xn2; pk.h[3] = (__bf16)xn3;
        *(uint2*)(xn_ws + (size_t)tok * DIM + tid * 4) = pk.u;
    }
    // out = x (residual base) and the x passthrough output
    ((float4*)(out_main  + (size_t)tok * DIM))[tid] = xv;
    ((float4*)(out_xcopy + (size_t)tok * DIM))[tid] = xv;

    // confidence dot product
    float4 cv = ((const float4*)conf_w)[tid];
    float csum = block_reduce_sum(xn0*cv.x + xn1*cv.y + xn2*cv.z + xn3*cv.w, red);

    // router logits
    float racc[NE];
#pragma unroll
    for (int e = 0; e < NE; ++e) racc[e] = 0.f;
    float xnv[4] = {xn0, xn1, xn2, xn3};
#pragma unroll
    for (int c = 0; c < 4; ++c) {
        int j = tid * 4 + c;
        float xj = xnv[c];
#pragma unroll
        for (int e = 0; e < NE; ++e) racc[e] += xj * rwt[e * DIM + j];
    }
#pragma unroll
    for (int e = 0; e < NE; ++e)
#pragma unroll
        for (int o = 16; o; o >>= 1) racc[e] += __shfl_xor(racc[e], o, 32);
    if (lane == 0) {
#pragma unroll
        for (int e = 0; e < NE; ++e) redE[w * NE + e] = racc[e];
    }
    __syncthreads();

    if (tid == 0) {
        float logit[NE];
        for (int e = 0; e < NE; ++e) {
            float t = router_b[e];
            for (int ww = 0; ww < 8; ++ww) t += redE[ww * NE + e];
            logit[e] = t;
        }
        float mx = logit[0];
        for (int e = 1; e < NE; ++e) mx = fmaxf(mx, logit[e]);
        float p[NE], psum = 0.f;
        for (int e = 0; e < NE; ++e) { p[e] = expf(logit[e] - mx); psum += p[e]; }
        float inv = 1.f / psum;
        for (int e = 0; e < NE; ++e) p[e] *= inv;
        int i0 = 0;
        for (int e = 1; e < NE; ++e) if (p[e] > p[i0]) i0 = e;   // ties -> lowest idx
        int i1 = (i0 == 0) ? 1 : 0;
        for (int e = 0; e < NE; ++e) if (e != i0 && p[e] > p[i1]) i1 = e;
        float g0 = p[i0] / (p[i0] + p[i1]);
        float g1 = p[i1] / (p[i0] + p[i1]);
        for (int e = 0; e < NE; ++e) cw[(size_t)tok * NE + e] = 0.f;
        cw[(size_t)tok * NE + i0] = g0;
        cw[(size_t)tok * NE + i1] = g1;
        int q0 = atomicAdd(&counts[i0], 1);
        lists[(size_t)i0 * N_TOK + q0] = tok;                    // slot 0 (top-1)
        int q1 = atomicAdd(&counts[NE + i1], 1);
        lists[(size_t)(NE + i1) * N_TOK + q1] = tok;             // slot 1 (top-2)
        out_conf[tok] = 1.f / (1.f + expf(-(csum + conf_b[0])));
    }
}

// --------------------------------------------- kernel 1.5: tile prefix sums
__global__ void prefix_kernel(const int* __restrict__ counts, int* __restrict__ tbase) {
    if (threadIdx.x == 0 && blockIdx.x == 0) {
        for (int s = 0; s < 2; ++s) {
            int acc = 0;
            for (int e = 0; e < NE; ++e) {
                tbase[s * 9 + e] = acc;
                acc += (counts[s * NE + e] + 15) >> 4;
            }
            tbase[s * 9 + NE] = acc;   // total tiles this slot
        }
    }
}

// ------------------------------------------------- kernel 2: expert FFN tile
__global__ __launch_bounds__(256) void moe_ffn_kernel(
    const __bf16* __restrict__ xn, const float* __restrict__ cw,
    const int* __restrict__ lists, const int* __restrict__ counts,
    const int* __restrict__ tbase,
    const float* __restrict__ w1, const float* __restrict__ b1,
    const float* __restrict__ w2, const float* __restrict__ b2,
    float* __restrict__ out)
{
    __shared__ __bf16 XN[16 * XN_STR];
    __shared__ __bf16 HC[16 * HC_STR];
    __shared__ int    toks[16];
    __shared__ float  gates[16];

    int b = blockIdx.x;
    if (b >= tbase[NE]) return;                     // uniform branch
    int e = 0;
    while (e < NE - 1 && b >= tbase[e + 1]) ++e;
    int t = b - tbase[e];
    int cnt = counts[e];

    int tid = threadIdx.x, lane = tid & 31, w = tid >> 5;

    if (tid < 16) {
        int idx = t * 16 + tid;
        int tok = (idx < cnt) ? lists[(size_t)e * N_TOK + idx] : -1;
        toks[tid]  = tok;
        gates[tid] = (tok >= 0) ? cw[(size_t)tok * NE + e] : 0.f;
    }
    __syncthreads();

    { // stage x_norm tile: 16 rows x 1024 cols, thread -> (row tid>>4, 64-col span)
        int r = tid >> 4, c0 = (tid & 15) * 64;
        int tok = toks[r];
        __bf16* dst = XN + r * XN_STR + c0;
        if (tok >= 0) {
            const __bf16* src = xn + (size_t)tok * DIM + c0;
#pragma unroll
            for (int i = 0; i < 16; ++i)
                *(uint2*)(dst + i * 4) = *(const uint2*)(src + i * 4);
        } else {
            uint2 z; z.x = 0u; z.y = 0u;
#pragma unroll
            for (int i = 0; i < 16; ++i) *(uint2*)(dst + i * 4) = z;
        }
    }
    __syncthreads();

    const float* w1e = w1 + (size_t)e * DIM * HID;   // [1024][4096]
    const float* b1e = b1 + (size_t)e * HID;
    const float* w2e = w2 + (size_t)e * HID * DIM;   // [4096][1024]
    const float* b2e = b2 + (size_t)e * DIM;

    v8f acc2[8] = {};                                // persistent out accumulators

    for (int hc = 0; hc < HID / HCHUNK; ++hc) {
        int hbase = hc * HCHUNK;
        // ---- phase 1: h = relu(xn @ W1 + b1) for 512 hidden cols
        v8f acc1[4] = {};
        for (int k0 = 0; k0 < DIM; k0 += 32) {
            v16bf a = load_a_frag(XN, XN_STR, k0);
#pragma unroll
            for (int tt = 0; tt < 4; ++tt) {
                v16bf bb = load_b_frag_global(w1e, HID, k0, hbase + w * 64 + tt * 16);
                acc1[tt] = wmma_bf16(a, bb, acc1[tt]);
            }
        }
#pragma unroll
        for (int tt = 0; tt < 4; ++tt) {
            int nloc = w * 64 + tt * 16 + (lane & 15);
            float bias = b1e[hbase + nloc];
            int half = lane >> 4;
#pragma unroll
            for (int j = 0; j < 8; ++j) {
                float v = acc1[tt][j] + bias;
                HC[(j + half * 8) * HC_STR + nloc] = (__bf16)fmaxf(v, 0.f);
            }
        }
        __syncthreads();
        // ---- phase 2: out += h_chunk @ W2[hbase:hbase+512, :]
        for (int k0 = 0; k0 < HCHUNK; k0 += 32) {
            v16bf a = load_a_frag(HC, HC_STR, k0);
#pragma unroll
            for (int tt = 0; tt < 8; ++tt) {
                v16bf bb = load_b_frag_global(w2e, DIM, hbase + k0, w * 128 + tt * 16);
                acc2[tt] = wmma_bf16(a, bb, acc2[tt]);
            }
        }
        __syncthreads();   // HC reused next chunk
    }

    // ---- epilogue: out[tok] += gate * (acc + b2)   (unique writer within slot pass)
#pragma unroll
    for (int tt = 0; tt < 8; ++tt) {
        int ncol = w * 128 + tt * 16 + (lane & 15);
        float bias = b2e[ncol];
        int half = lane >> 4;
#pragma unroll
        for (int j = 0; j < 8; ++j) {
            int row = j + half * 8;
            int tok = toks[row];
            if (tok >= 0) {
                float* p = out + (size_t)tok * DIM + ncol;
                *p += gates[row] * (acc2[tt][j] + bias);
            }
        }
    }
}

// ----------------------------------------------------------------- launcher
extern "C" void kernel_launch(void* const* d_in, const int* in_sizes, int n_in,
                              void* d_out, int out_size, void* d_ws, size_t ws_size,
                              hipStream_t stream) {
    const float* x        = (const float*)d_in[0];
    const float* ln_g     = (const float*)d_in[1];
    const float* ln_b     = (const float*)d_in[2];
    const float* conf_w   = (const float*)d_in[3];
    const float* conf_b   = (const float*)d_in[4];
    const float* router_w = (const float*)d_in[5];
    const float* router_b = (const float*)d_in[6];
    const float* w1       = (const float*)d_in[7];
    const float* b1       = (const float*)d_in[8];
    const float* w2       = (const float*)d_in[9];
    const float* b2       = (const float*)d_in[10];

    float* out_main  = (float*)d_out;
    float* out_conf  = out_main + (size_t)N_TOK * DIM;
    float* out_xcopy = out_conf + N_TOK;

    char* ws = (char*)d_ws;
    __bf16* xn  = (__bf16*)ws;            ws += (size_t)N_TOK * DIM * sizeof(__bf16);
    float* cw   = (float*)ws;             ws += (size_t)N_TOK * NE * sizeof(float);
    int* lists  = (int*)ws;               ws += (size_t)2 * NE * N_TOK * sizeof(int);
    int* counts = (int*)ws;               ws += 2 * NE * sizeof(int);
    int* tbase  = (int*)ws;               ws += 2 * 9 * sizeof(int);

    hipMemsetAsync(counts, 0, 2 * NE * sizeof(int), stream);

    ln_router_kernel<<<N_TOK, 256, 0, stream>>>(
        x, ln_g, ln_b, conf_w, conf_b, router_w, router_b,
        xn, cw, lists, counts, out_main, out_conf, out_xcopy);

    prefix_kernel<<<1, 32, 0, stream>>>(counts, tbase);

    // fixed grids (graph-capture safe); blocks beyond the live tile count exit early.
    const int MAX_TILES = N_TOK / 16 + NE;   // 520 >= worst-case sum of ceil(count/16)
    for (int slot = 0; slot < 2; ++slot) {
        moe_ffn_kernel<<<MAX_TILES, 256, 0, stream>>>(
            xn, cw,
            lists + (size_t)slot * NE * N_TOK,
            counts + slot * NE,
            tbase + slot * 9,
            w1, b1, w2, b2, out_main);
    }
}
